// RNNRegressor_6485400617694
// MI455X (gfx1250) — matrix operand order repaired and checked by reading, not compiled, not run on hardware
//
#include <hip/hip_runtime.h>
#include <hip/hip_bf16.h>
#include <math.h>
#include <stdint.h>

typedef _Float16 half_t;
typedef _Float16 v16h __attribute__((ext_vector_type(16)));
typedef _Float16 v8h  __attribute__((ext_vector_type(8)));
typedef float    v8f  __attribute__((ext_vector_type(8)));
typedef int      v4i  __attribute__((vector_size(16)));

typedef __attribute__((address_space(1))) v4i gvec_t;   // global <4 x i32>
typedef __attribute__((address_space(3))) v4i lvec_t;   // LDS    <4 x i32>

#define HDIM 160
#define BSZ  256
#define TLEN 2048
#define DIN  64
#define NT   10          // N tiles of 16 covering H=160
#define SH   168         // LDS h row stride in halves (padded)
#define PFD  8           // prefetch distance (timesteps)

// ---- async global->LDS copy of one 16B chunk (CDNA5 ASYNCcnt path) --------
static __device__ __forceinline__ void async_copy_b128(const half_t* gsrc, half_t* ldst) {
#if __has_builtin(__builtin_amdgcn_global_load_async_to_lds_b128)
  gvec_t* gp = (gvec_t*)(uintptr_t)gsrc;
  lvec_t* lp = (lvec_t*)(uint32_t)(uintptr_t)ldst;   // low 32b = LDS offset
  __builtin_amdgcn_global_load_async_to_lds_b128(gp, lp, 0, 0);
#else
  uint32_t lofs = (uint32_t)(uintptr_t)ldst;         // LDS byte address
  asm volatile("global_load_async_to_lds_b128 %0, %1, off"
               :: "v"(lofs), "v"(gsrc) : "memory");
#endif
}

static __device__ __forceinline__ void wait_asynccnt0() {
#if __has_builtin(__builtin_amdgcn_s_wait_asynccnt)
  __builtin_amdgcn_s_wait_asynccnt(0);
#else
  asm volatile("s_wait_asynccnt 0x0" ::: "memory");
#endif
}

// ---- WMMA fragment loaders (CDNA5 16x16x32 f16 layouts) --------------------

// A (16x32 f16) from LDS-resident h state. lane: M = lane&15, kb = (lane>>4)*8
// element e: k = 16*(e>>3) + kb + (e&7)  -> two contiguous 8-half (16B) groups.
static __device__ __forceinline__ v16h a_frag_lds(const half_t* hb, int lane, int kk) {
  int m  = lane & 15;
  int kb = (lane >> 4) * 8;
  const half_t* p = hb + m * SH + kk * 32 + kb;
  v8h lo = *(const v8h*)(p);
  v8h hi = *(const v8h*)(p + 16);
  v16h a;
#pragma unroll
  for (int i = 0; i < 8; ++i) { a[i] = lo[i]; a[8 + i] = hi[i]; }
  return a;
}

// A (16x32 f16) from a global f16 matrix with row stride ld.
static __device__ __forceinline__ v16h a_frag_f16(const half_t* A, size_t row0,
                                                  int ld, int lane, int kk) {
  int m  = lane & 15;
  int kb = (lane >> 4) * 8;
  const half_t* p = A + (row0 + m) * (size_t)ld + kk * 32 + kb;
  v8h lo = *(const v8h*)(p);
  v8h hi = *(const v8h*)(p + 16);
  v16h a;
#pragma unroll
  for (int i = 0; i < 8; ++i) { a[i] = lo[i]; a[8 + i] = hi[i]; }
  return a;
}

// A (16x32) from a global f32 matrix (convert to f16), row stride ld.
static __device__ __forceinline__ v16h a_frag_f32(const float* A, size_t row0,
                                                  int ld, int lane, int kk) {
  int m  = lane & 15;
  int kb = (lane >> 4) * 8;
  const float* p = A + (row0 + m) * (size_t)ld + kk * 32 + kb;
  v16h a;
#pragma unroll
  for (int i = 0; i < 8; ++i) { a[i] = (half_t)p[i]; a[8 + i] = (half_t)p[16 + i]; }
  return a;
}

// B (32x16 f16) tile of W^T from row-major W [N x K] (f32): B[k][j] = W[j][k].
// lane: j = n0 + (lane&15), k = kk*32 + 16*(e>>3) + 8*(lane>>4) + (e&7).
static __device__ __forceinline__ v16h b_frag_wT(const float* W, int ldk,
                                                 int n0, int lane, int kk) {
  int j = n0 + (lane & 15);
  const float* p = W + (size_t)j * ldk + kk * 32 + ((lane >> 4) * 8);
  v16h b;
#pragma unroll
  for (int e = 0; e < 8; ++e) { b[e] = (half_t)p[e]; b[8 + e] = (half_t)p[16 + e]; }
  return b;
}

// xp is stored time-major [T, B, H] so each timestep's workgroup tile
// (16 batch rows x 160 cols, f16) is one contiguous 5120B block.
static __device__ __forceinline__ void store_xp(half_t* xp, size_t row0, int mr0,
                                                int j, const v8f& c, float bias) {
  size_t bb = row0 / TLEN;
  int    t0 = (int)(row0 % TLEN);
#pragma unroll
  for (int v = 0; v < 8; ++v)
    xp[((size_t)(t0 + mr0 + v) * BSZ + bb) * HDIM + j] = (half_t)(c[v] + bias);
}

// ---- Kernel 1: xp0 = x @ Wih0^T + bih0 + bhh0  (f32 in, f16 [T,B,H] out) ---
__global__ void k_xproj0(const float* __restrict__ x, const float* __restrict__ Wih,
                         const float* __restrict__ bih, const float* __restrict__ bhh,
                         half_t* __restrict__ xp) {
  int lane = threadIdx.x & 31;
  int wv   = threadIdx.x >> 5;                       // 8 waves / block
  size_t row0 = ((size_t)blockIdx.x * 8 + wv) * 16;  // M tile over B*T (b*T+t)
  int n0 = blockIdx.y * 16;

  v16h bf[2];
#pragma unroll
  for (int kk = 0; kk < 2; ++kk) bf[kk] = b_frag_wT(Wih, DIN, n0, lane, kk);

  v8f c = {};
#pragma unroll
  for (int kk = 0; kk < 2; ++kk) {
    v16h a = a_frag_f32(x, row0, DIN, lane, kk);
    c = __builtin_amdgcn_wmma_f32_16x16x32_f16(false, a, false, bf[kk],
                                               (short)0, c, false, false);
  }
  int j = n0 + (lane & 15);
  store_xp(xp, row0, 8 * (lane >> 4), j, c, bih[j] + bhh[j]);
}

// ---- Kernel 3: xp1 = y1 @ Wih1^T + bih1 + bhh1  (f16 in, f16 [T,B,H] out) --
__global__ void k_xproj1(const half_t* __restrict__ y1, const float* __restrict__ Wih,
                         const float* __restrict__ bih, const float* __restrict__ bhh,
                         half_t* __restrict__ xp) {
  int lane = threadIdx.x & 31;
  int wv   = threadIdx.x >> 5;
  size_t row0 = ((size_t)blockIdx.x * 8 + wv) * 16;
  int n0 = blockIdx.y * 16;

  v16h bf[5];
#pragma unroll
  for (int kk = 0; kk < 5; ++kk) bf[kk] = b_frag_wT(Wih, HDIM, n0, lane, kk);

  v8f c = {};
#pragma unroll
  for (int kk = 0; kk < 5; ++kk) {
    v16h a = a_frag_f16(y1, row0, HDIM, lane, kk);
    c = __builtin_amdgcn_wmma_f32_16x16x32_f16(false, a, false, bf[kk],
                                               (short)0, c, false, false);
  }
  int j = n0 + (lane & 15);
  store_xp(xp, row0, 8 * (lane >> 4), j, c, bih[j] + bhh[j]);
}

// ---- Recurrence: h_t = tanh(xp_t + h_{t-1} @ Whh^T) ------------------------
// One block per 16-row batch tile; 10 waves, one per 16-col N tile of H.
// h state lives in LDS (f16); Whh^T tiles resident in VGPRs.
// The xp tile for each step (contiguous 5120B in [T,B,H] layout) is copied
// into a double-buffered LDS tile by the async DMA path (ASYNCcnt-tracked
// global_load_async_to_lds_b128, 320 threads x 16B), one step ahead, with
// global_prefetch_b8 issued PFD steps ahead to keep the stream L2-resident.
template <bool COLLECT>
__global__ void k_rnn(const half_t* __restrict__ xp, const float* __restrict__ Whh,
                      half_t* __restrict__ yout, float* __restrict__ hfin) {
  __shared__ __align__(16) half_t hb[16 * SH];
  __shared__ __align__(16) half_t xt[2][16 * HDIM];
  int tid  = threadIdx.x;               // 0..319
  int lane = tid & 31;
  int wv   = tid >> 5;                  // 0..9 -> N tile
  int b0   = blockIdx.x * 16;           // batch tile
  int n0   = wv * 16;
  int j    = n0 + (lane & 15);
  int mr0  = 8 * (lane >> 4);

  v16h bf[5];
#pragma unroll
  for (int kk = 0; kk < 5; ++kk) bf[kk] = b_frag_wT(Whh, HDIM, n0, lane, kk);

  for (int i = tid; i < 16 * SH; i += blockDim.x) hb[i] = (half_t)0.f;

  // async-stage timestep 0 tile
  const half_t* xtile0  = xp + (size_t)b0 * HDIM;   // [T, B, H] time-major
  const size_t  tstride = (size_t)BSZ * HDIM;       // halves per timestep
  async_copy_b128(xtile0 + (size_t)tid * 8, &xt[0][tid * 8]);
  wait_asynccnt0();
  __syncthreads();

  for (int t = 0; t < TLEN; ++t) {
    int buf = t & 1;
    // kick off async copy of the next step's tile into the other LDS buffer
    if (t + 1 < TLEN)
      async_copy_b128(xtile0 + (size_t)(t + 1) * tstride + (size_t)tid * 8,
                      &xt[buf ^ 1][tid * 8]);
    // prefetch the xp stream PFD steps ahead into cache
    if (t + PFD < TLEN)
      __builtin_prefetch(xtile0 + (size_t)(t + PFD) * tstride + (size_t)tid * 8, 0, 0);

    // accumulator initialized from LDS-staged input projection
    v8f c;
#pragma unroll
    for (int v = 0; v < 8; ++v)
      c[v] = (float)xt[buf][(mr0 + v) * HDIM + j];

#pragma unroll
    for (int kk = 0; kk < 5; ++kk) {
      v16h a = a_frag_lds(hb, lane, kk);
      c = __builtin_amdgcn_wmma_f32_16x16x32_f16(false, a, false, bf[kk],
                                                 (short)0, c, false, false);
    }
#pragma unroll
    for (int v = 0; v < 8; ++v) c[v] = tanhf(c[v]);

    __syncthreads();   // all waves done reading h_{t-1} / xt[buf]
#pragma unroll
    for (int v = 0; v < 8; ++v) {
      half_t hv = (half_t)c[v];
      hb[(mr0 + v) * SH + j] = hv;
      if (COLLECT)
        yout[((size_t)(b0 + mr0 + v) * TLEN + t) * HDIM + j] = hv;
    }
    if (t + 1 < TLEN) wait_asynccnt0();   // this wave's async tile landed
    __syncthreads();   // h_t written + all waves' async tiles visible

    if (!COLLECT && t == TLEN - 1) {
#pragma unroll
      for (int v = 0; v < 8; ++v)
        hfin[(size_t)(b0 + mr0 + v) * HDIM + j] = c[v];
    }
  }
}

// ---- Head: out[b] = h2[b,:] . W_head + b_head ------------------------------
__global__ void k_head(const float* __restrict__ h2, const float* __restrict__ Wh,
                       const float* __restrict__ bh, float* __restrict__ out) {
  int b = blockIdx.x * blockDim.x + threadIdx.x;
  if (b >= BSZ) return;
  float s = bh[0];
#pragma unroll 4
  for (int jj = 0; jj < HDIM; ++jj) s += h2[(size_t)b * HDIM + jj] * Wh[jj];
  out[b] = s;
}

extern "C" void kernel_launch(void* const* d_in, const int* in_sizes, int n_in,
                              void* d_out, int out_size, void* d_ws, size_t ws_size,
                              hipStream_t stream) {
  (void)in_sizes; (void)n_in; (void)out_size; (void)ws_size;
  const float* x     = (const float*)d_in[0];
  const float* Wih0  = (const float*)d_in[1];
  const float* Whh0  = (const float*)d_in[2];
  const float* bih0  = (const float*)d_in[3];
  const float* bhh0  = (const float*)d_in[4];
  const float* Wih1  = (const float*)d_in[5];
  const float* Whh1  = (const float*)d_in[6];
  const float* bih1  = (const float*)d_in[7];
  const float* bhh1  = (const float*)d_in[8];
  const float* Whead = (const float*)d_in[9];
  const float* bhead = (const float*)d_in[10];
  float* out = (float*)d_out;

  const size_t XP_BYTES = (size_t)BSZ * TLEN * HDIM * sizeof(half_t); // 160 MiB
  char* ws = (char*)d_ws;
  half_t* xp = (half_t*)(ws);                 // [T, B, H] f16 (reused per layer)
  half_t* y1 = (half_t*)(ws + XP_BYTES);      // [B, T, H] f16
  float*  h2 = (float*)(ws + 2 * XP_BYTES);   // [B, H] f32

  dim3 gGemm((BSZ * TLEN) / (16 * 8), NT);    // 4096 x 10, 8 waves/block

  // layer 0
  k_xproj0<<<gGemm, 256, 0, stream>>>(x, Wih0, bih0, bhh0, xp);
  k_rnn<true><<<BSZ / 16, NT * 32, 0, stream>>>(xp, Whh0, y1, nullptr);
  // layer 1
  k_xproj1<<<gGemm, 256, 0, stream>>>(y1, Wih1, bih1, bhh1, xp);
  k_rnn<false><<<BSZ / 16, NT * 32, 0, stream>>>(xp, Whh1, nullptr, h2);
  // head
  k_head<<<1, 256, 0, stream>>>(h2, Whead, bhead, out);
}
